// RelationalEncoder_50337016709886
// MI455X (gfx1250) — matrix operand order repaired and checked by reading, not compiled, not run on hardware
//
#include <hip/hip_runtime.h>
#include <cstdint>
#include <cstddef>

// ---------------------------------------------------------------------------
// Problem constants (match reference)
// ---------------------------------------------------------------------------
constexpr int B_ = 64, N_ = 64, T_ = 128;
constexpr int E_ = N_ * (N_ - 1);          // 4032
constexpr int RN = B_ * N_;                // 4096 node rows
constexpr int RE = B_ * E_;                // 258048 edge rows
constexpr float EPS = 1e-5f;

typedef __attribute__((ext_vector_type(16))) __bf16        v16bf;
typedef __attribute__((ext_vector_type(8)))  float         v8f;
typedef __attribute__((ext_vector_type(8)))  unsigned int  v8u;
typedef __attribute__((ext_vector_type(4)))  unsigned int  v4u;

#define DEV __device__ __forceinline__

DEV unsigned short f2bf(float f) {                 // round-to-nearest-even
  union { float f; unsigned u; } x; x.f = f;
  unsigned r = x.u + 0x7FFFu + ((x.u >> 16) & 1u);
  return (unsigned short)(r >> 16);
}
DEV float bf2f(unsigned short h) {
  union { unsigned u; float f; } x; x.u = ((unsigned)h) << 16;
  return x.f;
}
DEV float elu(float v) {                           // hw transcendental, no libm
  return v > 0.f ? v : (__expf(v) - 1.0f);
}

// ---------------------------------------------------------------------------
// Small prep kernels
// ---------------------------------------------------------------------------
__global__ void cvt_f32_bf16(const float* __restrict__ in,
                             unsigned short* __restrict__ out, int n) {
  int i = blockIdx.x * 256 + threadIdx.x;
  if (i < n) out[i] = f2bf(in[i]);
}

// Pack f32 weight [K,256] into WMMA-B fragment order (bf16):
//   dword index = ((ct*KT + kt)*32 + lane)*8 + t
//   lane n = ct*16 + (lane&15), kbase = (lane>>4)*16, k = kt*32 + kbase + 2t
// A lane's whole fragment = 8 consecutive dwords = one 32B vector load.
__global__ void pack_w(const float* __restrict__ w,
                       unsigned short* __restrict__ wp, int K) {
  int idx = blockIdx.x * 256 + threadIdx.x;
  int KT = K >> 5;
  if (idx >= 16 * KT * 32 * 8) return;
  int t    = idx & 7;
  int lane = (idx >> 3) & 31;
  int kt   = (idx >> 8) % KT;
  int ct   = (idx >> 8) / KT;
  int n = ct * 16 + (lane & 15);
  int k = kt * 32 + ((lane >> 4) << 4) + 2 * t;
  unsigned int lo = f2bf(w[(size_t)k * 256 + n]);
  unsigned int hi = f2bf(w[(size_t)(k + 1) * 256 + n]);
  ((unsigned int*)wp)[idx] = lo | (hi << 16);
}

// rel_rec / rel_send are one-hot [E,N] -> extract column index per edge
__global__ void onehot_idx(const float* __restrict__ rr,
                           const float* __restrict__ rs,
                           int* __restrict__ recv_e, int* __restrict__ send_e) {
  int e = blockIdx.x * 256 + threadIdx.x;
  if (e >= E_) return;
  int r = 0, s = 0;
  for (int n = 0; n < N_; ++n) {
    if (rr[e * N_ + n] > 0.5f) r = n;
    if (rs[e * N_ + n] > 0.5f) s = n;
  }
  recv_e[e] = r; send_e[e] = s;
}

// Row-index maps for fused gather: global edge-row -> node-row
__global__ void expand_rows(const int* __restrict__ recv_e,
                            const int* __restrict__ send_e,
                            int* __restrict__ rridx, int* __restrict__ sridx) {
  int i = blockIdx.x * 256 + threadIdx.x;
  if (i >= RE) return;
  int b = i / E_, e = i - b * E_;
  rridx[i] = b * N_ + recv_e[e];
  sridx[i] = b * N_ + send_e[e];
}

// Deterministic incoming-edge list per node (63 edges each)
__global__ void build_inlist(const int* __restrict__ recv_e, int* __restrict__ inlist) {
  int n = threadIdx.x;             // 64 threads
  if (n >= N_) return;
  int c = 0;
  for (int e = 0; e < E_; ++e)
    if (recv_e[e] == n) inlist[n * (N_ - 1) + (c++)] = e;
}

// ---------------------------------------------------------------------------
// Core GEMM: C[rows,256] = elu( A[rows,K] * W[K,256] + bias )
//   * A from up to 3 parts (K offsets 0/256/512) with optional row-index
//     indirection (fused node2edge gather / skip concat), staged via LDS
//     with b128 copies (16B-aligned padded stride).
//   * B read directly from pre-packed global (one v8u per lane per tile).
//   * optional fused per-feature sum / sum^2 for BatchNorm statistics.
// Block: 256 thr = 8 wave32; tile 128(M) x 128(N); each wave: 2 strips of
// 16x64 -> 8 x v_wmma_f32_16x16x32_bf16 per 32-K step. grid = (rows/128, 2)
// ---------------------------------------------------------------------------
#define ASTR 40   // halves; 80B row stride: 16B aligned, conflict-free frags

__global__ __launch_bounds__(256)
void gemm_bf16(const unsigned short* __restrict__ A0, const int* __restrict__ I0, int lda0,
               const unsigned short* __restrict__ A1, const int* __restrict__ I1, int lda1,
               const unsigned short* __restrict__ A2, const int* __restrict__ I2, int lda2,
               const unsigned short* __restrict__ Wp, const float* __restrict__ bias,
               unsigned short* __restrict__ outb,
               float* __restrict__ gsum, float* __restrict__ gsq,
               int rows, int Ktotal) {
  __shared__ __align__(16) unsigned short As[128 * ASTR];
  __shared__ int   rowmap[3 * 128];
  __shared__ float lsum[128], lsq[128];

  const int tid  = threadIdx.x;
  const int row0 = blockIdx.x * 128;
  const int col0 = blockIdx.y * 128;
  const int KT   = Ktotal >> 5;

  for (int q = tid; q < 3 * 128; q += 256) {   // per-part row maps
    int p = q >> 7, r = q & 127;
    int g = row0 + r; if (g >= rows) g = rows - 1;
    const int* idx = (p == 0) ? I0 : (p == 1) ? I1 : I2;
    rowmap[q] = idx ? idx[g] : g;
  }
  if (gsum && tid < 128) { lsum[tid] = 0.f; lsq[tid] = 0.f; }
  __syncthreads();

  const int lane = tid & 31, wv = tid >> 5;
  const int wm = (wv & 3) * 16;          // strip rows: wm and wm+64
  const int wn = (wv >> 2) * 64;         // wave col offset inside tile
  const int ctbase = (col0 >> 4) + (wn >> 4);

  const v8f vz = {0.f, 0.f, 0.f, 0.f, 0.f, 0.f, 0.f, 0.f};
  v8f acc[2][4] = {{vz, vz, vz, vz}, {vz, vz, vz, vz}};
  const v8u* __restrict__ Wv = (const v8u*)Wp;

  for (int k0 = 0; k0 < Ktotal; k0 += 32) {
    const int p = k0 >> 8;               // parts are 256-wide (128 for MLP1)
    const unsigned short* Ab = (p == 0) ? A0 : (p == 1) ? A1 : A2;
    const int lda = (p == 0) ? lda0 : (p == 1) ? lda1 : lda2;
    const int kk = k0 - (p << 8);

    // ---- stage A tile 128x32: 512 16B chunks, 2 x b128 per thread ----
    #pragma unroll
    for (int i = 0; i < 2; ++i) {
      int q = tid + (i << 8);
      int r = q >> 2, c = q & 3;         // row, 16B-chunk (8 halves)
      v4u v = *(const v4u*)(Ab + (size_t)rowmap[p * 128 + r] * lda + kk + c * 8);
      *(v4u*)(&As[r * ASTR + c * 8]) = v;
    }
    __syncthreads();

    // ---- B fragments: direct 32B vector loads from packed weights ----
    const int kt = k0 >> 5;
    v16bf bf[4];
    #pragma unroll
    for (int j = 0; j < 4; ++j)
      bf[j] = __builtin_bit_cast(v16bf,
                Wv[((size_t)(ctbase + j) * KT + kt) * 32 + lane]);

    // ---- A fragments: two contiguous 16B LDS loads per lane + 8 WMMA ----
    const int kbA = (lane >> 4) * 16;    // halves: K kbase..kbase+15
    #pragma unroll
    for (int s = 0; s < 2; ++s) {
      const int m = s * 64 + wm + (lane & 15);
      union { v8u u8; v4u u4[2]; } au;
      au.u4[0] = *(const v4u*)(&As[m * ASTR + kbA]);
      au.u4[1] = *(const v4u*)(&As[m * ASTR + 8 + kbA]);
      v16bf af = __builtin_bit_cast(v16bf, au.u8);
      #pragma unroll
      for (int j = 0; j < 4; ++j)
        acc[s][j] = __builtin_amdgcn_wmma_f32_16x16x32_bf16(
            false, af, false, bf[j], (short)0, acc[s][j], false, false);
    }
    __syncthreads();
  }

  // ---- epilogue: bias + ELU + store bf16 (+ fused BN statistics) ----
  const int mofs = wm + ((lane >> 4) ? 8 : 0);   // C layout: vgpr i -> M=i(+8)
  #pragma unroll
  for (int j = 0; j < 4; ++j) {
    int lc  = wn + j * 16 + (lane & 15);
    int col = col0 + lc;
    float bv = bias[col];
    float cs = 0.f, cq = 0.f;
    #pragma unroll
    for (int s = 0; s < 2; ++s) {
      #pragma unroll
      for (int i = 0; i < 8; ++i) {
        float v = elu(acc[s][j][i] + bv);
        int row = row0 + s * 64 + mofs + i;
        outb[(size_t)row * 256 + col] = f2bf(v);
        cs += v; cq += v * v;
      }
    }
    if (gsum) { atomicAdd(&lsum[lc], cs); atomicAdd(&lsq[lc], cq); }
  }
  if (gsum) {
    __syncthreads();
    if (tid < 128) {
      unsafeAtomicAdd(&gsum[col0 + tid], lsum[tid]);
      unsafeAtomicAdd(&gsq [col0 + tid], lsq [tid]);
    }
  }
}

// ---------------------------------------------------------------------------
// BatchNorm finalize + apply
// ---------------------------------------------------------------------------
__global__ void bn_finalize(const float* __restrict__ sum, const float* __restrict__ sq,
                            float inv_r, const float* __restrict__ g,
                            const float* __restrict__ be,
                            float* __restrict__ scale, float* __restrict__ shift) {
  int f = threadIdx.x;                   // 256
  float mu  = sum[f] * inv_r;
  float var = sq[f] * inv_r - mu * mu;
  float sc  = g[f] * rsqrtf(var + EPS);
  scale[f] = sc;
  shift[f] = be[f] - mu * sc;
}

__global__ void bn_apply(const unsigned short* __restrict__ h,
                         const float* __restrict__ scale, const float* __restrict__ shift,
                         unsigned short* __restrict__ outb, float* __restrict__ outf) {
  size_t i = (size_t)blockIdx.x * 256 + threadIdx.x;   // one row per block
  int f = threadIdx.x;
  float v = bf2f(h[i]) * scale[f] + shift[f];
  if (outb) outb[i] = f2bf(v);
  if (outf) outf[i] = v;
}

// edge2node: x_node[b,n,:] = (1/N) * sum_{e: recv[e]==n} x_edge[b,e,:]
__global__ void edge2node(const unsigned short* __restrict__ xe,
                          const int* __restrict__ inlist,
                          unsigned short* __restrict__ xn) {
  int blk = blockIdx.x;                  // RN blocks
  int b = blk >> 6, n = blk & 63;
  int f = threadIdx.x;
  float acc = 0.f;
  for (int j = 0; j < N_ - 1; ++j) {
    int e = inlist[n * (N_ - 1) + j];
    acc += bf2f(xe[((size_t)(b * E_ + e)) * 256 + f]);
  }
  xn[(size_t)blk * 256 + f] = f2bf(acc * (1.0f / (float)N_));
}

// ---------------------------------------------------------------------------
// Host orchestration
// ---------------------------------------------------------------------------
extern "C" void kernel_launch(void* const* d_in, const int* in_sizes, int n_in,
                              void* d_out, int out_size, void* d_ws, size_t ws_size,
                              hipStream_t stream) {
  (void)in_sizes; (void)n_in; (void)out_size; (void)ws_size;

  const float* inputs  = (const float*)d_in[0];
  const float* rel_rec = (const float*)d_in[1];
  const float* rel_snd = (const float*)d_in[2];
  auto P = [&](int i, int slot) { return (const float*)d_in[3 + (i - 1) * 6 + slot]; };
  // slot: 0=w1 1=b1 2=w2 3=b2 4=g 5=be

  char* ws = (char*)d_ws;
  size_t off = 0;
  auto alloc = [&](size_t bytes) -> char* {
    char* p = ws + off;
    off += (bytes + 255) & ~(size_t)255;
    return p;
  };

  unsigned short* xin = (unsigned short*)alloc((size_t)RN * T_ * 2);
  const int wk[10] = {128, 256, 512, 256, 256, 256, 768, 256, 256, 256};
  unsigned short* wbf[10];
  for (int i = 0; i < 10; ++i) wbf[i] = (unsigned short*)alloc((size_t)wk[i] * 256 * 2);

  int* recv_e = (int*)alloc(E_ * 4);
  int* send_e = (int*)alloc(E_ * 4);
  int* rridx  = (int*)alloc((size_t)RE * 4);
  int* sridx  = (int*)alloc((size_t)RE * 4);
  int* inlist = (int*)alloc(N_ * (N_ - 1) * 4);
  float* gsum  = (float*)alloc(256 * 4);
  float* gsq   = (float*)alloc(256 * 4);
  float* scale = (float*)alloc(256 * 4);
  float* shift = (float*)alloc(256 * 4);

  unsigned short* h_node  = (unsigned short*)alloc((size_t)RN * 256 * 2);
  unsigned short* h2_node = (unsigned short*)alloc((size_t)RN * 256 * 2);
  unsigned short* x1bn    = (unsigned short*)alloc((size_t)RN * 256 * 2);
  unsigned short* xnodes  = (unsigned short*)alloc((size_t)RN * 256 * 2);
  unsigned short* x3bn    = (unsigned short*)alloc((size_t)RN * 256 * 2);
  unsigned short* h_edge  = (unsigned short*)alloc((size_t)RE * 256 * 2);
  unsigned short* h2_edge = (unsigned short*)alloc((size_t)RE * 256 * 2);
  unsigned short* x2bn    = (unsigned short*)alloc((size_t)RE * 256 * 2);
  unsigned short* x4bn    = x2bn;  // x2bn dead after MLP4 layer-1 -> alias

  // ---- prep: input conversion, weight packing, graph index extraction ----
  cvt_f32_bf16<<<(RN * T_ + 255) / 256, 256, 0, stream>>>(inputs, xin, RN * T_);
  for (int i = 0; i < 5; ++i) {
    int k1 = wk[2 * i], k2 = wk[2 * i + 1];
    pack_w<<<(16 * (k1 >> 5) * 256 + 255) / 256, 256, 0, stream>>>(P(i + 1, 0), wbf[2 * i], k1);
    pack_w<<<(16 * (k2 >> 5) * 256 + 255) / 256, 256, 0, stream>>>(P(i + 1, 2), wbf[2 * i + 1], k2);
  }
  onehot_idx<<<(E_ + 255) / 256, 256, 0, stream>>>(rel_rec, rel_snd, recv_e, send_e);
  expand_rows<<<(RE + 255) / 256, 256, 0, stream>>>(recv_e, send_e, rridx, sridx);
  build_inlist<<<1, 64, 0, stream>>>(recv_e, inlist);

  auto gemm = [&](const unsigned short* A0, const int* I0, int lda0,
                  const unsigned short* A1, const int* I1, int lda1,
                  const unsigned short* A2, const int* I2, int lda2,
                  const unsigned short* Wm, const float* bias,
                  unsigned short* ob, float* gs, float* gq, int rows, int Kt) {
    dim3 grid(rows / 128, 2);
    gemm_bf16<<<grid, 256, 0, stream>>>(A0, I0, lda0, A1, I1, lda1, A2, I2, lda2,
                                        Wm, bias, ob, gs, gq, rows, Kt);
  };
  auto zero_stats = [&]() {
    hipMemsetAsync(gsum, 0, 256 * 4, stream);
    hipMemsetAsync(gsq, 0, 256 * 4, stream);
  };

  // ---- MLP1: [B*N,128] -> x1bn [B*N,256] ----
  gemm(xin, nullptr, T_, xin, nullptr, T_, xin, nullptr, T_,
       wbf[0], P(1, 1), h_node, nullptr, nullptr, RN, T_);
  zero_stats();
  gemm(h_node, nullptr, 256, h_node, nullptr, 256, h_node, nullptr, 256,
       wbf[1], P(1, 3), h2_node, gsum, gsq, RN, 256);
  bn_finalize<<<1, 256, 0, stream>>>(gsum, gsq, 1.0f / RN, P(1, 4), P(1, 5), scale, shift);
  bn_apply<<<RN, 256, 0, stream>>>(h2_node, scale, shift, x1bn, nullptr);

  // ---- MLP2: gather-fused [B*E,512] -> x2bn [B*E,256] ----
  gemm(x1bn, rridx, 256, x1bn, sridx, 256, x1bn, nullptr, 256,
       wbf[2], P(2, 1), h_edge, nullptr, nullptr, RE, 512);
  zero_stats();
  gemm(h_edge, nullptr, 256, h_edge, nullptr, 256, h_edge, nullptr, 256,
       wbf[3], P(2, 3), h2_edge, gsum, gsq, RE, 256);
  bn_finalize<<<1, 256, 0, stream>>>(gsum, gsq, 1.0f / RE, P(2, 4), P(2, 5), scale, shift);
  bn_apply<<<RE, 256, 0, stream>>>(h2_edge, scale, shift, x2bn, nullptr);

  // ---- edge2node ----
  edge2node<<<RN, 256, 0, stream>>>(x2bn, inlist, xnodes);

  // ---- MLP3: [B*N,256] -> x3bn ----
  gemm(xnodes, nullptr, 256, xnodes, nullptr, 256, xnodes, nullptr, 256,
       wbf[4], P(3, 1), h_node, nullptr, nullptr, RN, 256);
  zero_stats();
  gemm(h_node, nullptr, 256, h_node, nullptr, 256, h_node, nullptr, 256,
       wbf[5], P(3, 3), h2_node, gsum, gsq, RN, 256);
  bn_finalize<<<1, 256, 0, stream>>>(gsum, gsq, 1.0f / RN, P(3, 4), P(3, 5), scale, shift);
  bn_apply<<<RN, 256, 0, stream>>>(h2_node, scale, shift, x3bn, nullptr);

  // ---- MLP4: gather+skip fused [B*E,768] -> x4bn ----
  gemm(x3bn, rridx, 256, x3bn, sridx, 256, x2bn, nullptr, 256,
       wbf[6], P(4, 1), h_edge, nullptr, nullptr, RE, 768);
  zero_stats();
  gemm(h_edge, nullptr, 256, h_edge, nullptr, 256, h_edge, nullptr, 256,
       wbf[7], P(4, 3), h2_edge, gsum, gsq, RE, 256);
  bn_finalize<<<1, 256, 0, stream>>>(gsum, gsq, 1.0f / RE, P(4, 4), P(4, 5), scale, shift);
  bn_apply<<<RE, 256, 0, stream>>>(h2_edge, scale, shift, x4bn, nullptr);

  // ---- MLP5: [B*E,256] -> d_out (f32) ----
  gemm(x4bn, nullptr, 256, x4bn, nullptr, 256, x4bn, nullptr, 256,
       wbf[8], P(5, 1), h_edge, nullptr, nullptr, RE, 256);
  zero_stats();
  gemm(h_edge, nullptr, 256, h_edge, nullptr, 256, h_edge, nullptr, 256,
       wbf[9], P(5, 3), h2_edge, gsum, gsq, RE, 256);
  bn_finalize<<<1, 256, 0, stream>>>(gsum, gsq, 1.0f / RE, P(5, 4), P(5, 5), scale, shift);
  bn_apply<<<RE, 256, 0, stream>>>(h2_edge, scale, shift, nullptr, (float*)d_out);
}
